// SCIF_Z_Model_28140625723886
// MI455X (gfx1250) — matrix operand order repaired and checked by reading, not compiled, not run on hardware
//
#include <hip/hip_runtime.h>
#include <hip/hip_bf16.h>
#include <math.h>

// ---------------------------------------------------------------------------
// Types for CDNA5 WMMA (wave32, v_wmma_f32_16x16x32_f16)
// ---------------------------------------------------------------------------
typedef __attribute__((ext_vector_type(16))) _Float16 v16h;
typedef __attribute__((ext_vector_type(8)))  _Float16 v8h;
typedef __attribute__((ext_vector_type(4)))  _Float16 v4h;
typedef __attribute__((ext_vector_type(8)))  float    v8f;

#define B_ROWS 8192
#define TM 128
#define TN 128
#define TK 32

__device__ __forceinline__ float gelu_f(float x) {
    return 0.5f * x * (1.0f + erff(x * 0.70710678118654752f));
}
__device__ __forceinline__ float dgelu_f(float x) {
    float c = 0.5f * (1.0f + erff(x * 0.70710678118654752f));
    return c + x * 0.3989422804014327f * __expf(-0.5f * x * x);
}
__device__ __forceinline__ float wave_sum(float v) {
    #pragma unroll
    for (int off = 16; off > 0; off >>= 1) v += __shfl_xor(v, off, 32);
    return v;
}
__device__ __forceinline__ v4h cvt4(float4 f) {
    v4h h; h[0] = (_Float16)f.x; h[1] = (_Float16)f.y;
    h[2] = (_Float16)f.z; h[3] = (_Float16)f.w; return h;
}

// ---------------------------------------------------------------------------
// Generic WMMA GEMM:  C[M,N] = A[M,K] @ B + bias, with epilogue options.
// transB==0: B is row-major KxN.   transB==1: memory holds W row-major NxK
// (so we compute A @ W^T).  All call sites: M%128==0, N%128==0, K%32==0.
// epilogue: 0 = bias only; 1 = gelu(bias+acc); 2 = C=z, C2=gelu(z).
// Block: 256 threads = 8 waves (4 x 2). Wave tile 32x64 = 2x4 WMMA frags.
// Per K-slab: 8 v_wmma per wave fed by 12 ds_load_b128.
// Double-buffered LDS (32KB); next-slab global_load_b128 issued before the
// WMMA phase so HBM latency hides behind the matrix pipe; one barrier/slab.
// ---------------------------------------------------------------------------
__global__ __launch_bounds__(256)
void gemm_wmma_kernel(const float* __restrict__ A, const float* __restrict__ Bm,
                      const float* __restrict__ bias, float* __restrict__ C,
                      float* __restrict__ C2, int M, int N, int K,
                      int transB, int epilogue)
{
    __shared__ _Float16 As[2][TM * TK];   // [m][k]
    __shared__ _Float16 Bs[2][TN * TK];   // [n][k] (N-major: B-frag reads contiguous)

    const int tid   = threadIdx.x;
    const int lane  = tid & 31;
    const int wave  = tid >> 5;
    const int waveM = wave & 3;           // 4 waves along M (32 rows each)
    const int waveN = wave >> 2;          // 2 waves along N (64 cols each)
    const int tileM = blockIdx.y * TM;
    const int tileN = blockIdx.x * TN;

    // ---- staging geometry: 4 chunks/thread, chunk offsets are wave-uniform ----
    const int aRow = tid >> 3;            // 0..31 (chunk i adds 32 rows)
    const int aCol = (tid & 7) << 2;      // 0,4,...,28
    const float* aCur = A + (size_t)(tileM + aRow) * K + aCol;
    const size_t aChunk = (size_t)32 * K;
    const int aDstBase = aRow * TK + aCol;

    const float* bCur;
    size_t bChunk, bAdv;
    int bDstBase, bDstStride;
    if (transB) {
        const int bRow = tid >> 3, bCol = (tid & 7) << 2;
        bCur = Bm + (size_t)(tileN + bRow) * K + bCol;
        bChunk = (size_t)32 * K;
        bAdv = TK;
        bDstBase = bRow * TK + bCol; bDstStride = 32 * TK;
    } else {
        const int bK = tid >> 5, bN = (tid & 31) << 2;
        bCur = Bm + (size_t)bK * N + tileN + bN;
        bChunk = (size_t)8 * N;
        bAdv = (size_t)TK * N;
        bDstBase = bN * TK + bK; bDstStride = 8;
    }

    v8f acc[2][4];
    #pragma unroll
    for (int mi = 0; mi < 2; ++mi)
        #pragma unroll
        for (int f = 0; f < 4; ++f)
            #pragma unroll
            for (int r = 0; r < 8; ++r) acc[mi][f][r] = 0.0f;

    const int lm = lane & 15;
    const int kh = (lane >> 4) << 3;      // A frag K offset: 0 or 8
    const int kb = (lane >> 4) << 3;      // B frag K offset: 0 or 8

    float4 av[4], bv[4];
    auto loadG = [&]() {
        #pragma unroll
        for (int i = 0; i < 4; ++i) av[i] = *(const float4*)(aCur + aChunk * i);
        #pragma unroll
        for (int i = 0; i < 4; ++i) bv[i] = *(const float4*)(bCur + bChunk * i);
        aCur += TK; bCur += bAdv;
        __builtin_prefetch((const void*)aCur, 0, 1);   // global_prefetch_b8
        __builtin_prefetch((const void*)bCur, 0, 1);
    };
    auto storeL = [&](int buf) {
        #pragma unroll
        for (int i = 0; i < 4; ++i)
            *(v4h*)&As[buf][aDstBase + 1024 * i] = cvt4(av[i]);
        if (transB) {
            #pragma unroll
            for (int i = 0; i < 4; ++i)
                *(v4h*)&Bs[buf][bDstBase + bDstStride * i] = cvt4(bv[i]);
        } else {
            #pragma unroll
            for (int i = 0; i < 4; ++i) {             // scatter: Bs is n-major
                _Float16* d = &Bs[buf][bDstBase + bDstStride * i];
                d[0 * TK] = (_Float16)bv[i].x;
                d[1 * TK] = (_Float16)bv[i].y;
                d[2 * TK] = (_Float16)bv[i].z;
                d[3 * TK] = (_Float16)bv[i].w;
            }
        }
    };
    auto compute = [&](int buf) {
        union { v16h v; v8h h[2]; } af[2];
        #pragma unroll
        for (int mi = 0; mi < 2; ++mi) {
            int m = (waveM << 5) + (mi << 4) + lm;
            af[mi].h[0] = *(const v8h*)&As[buf][m * TK + kh];
            af[mi].h[1] = *(const v8h*)&As[buf][m * TK + kh + 16];
        }
        #pragma unroll
        for (int f = 0; f < 4; ++f) {
            union { v16h v; v8h h[2]; } bf;
            int n = (waveN << 6) + (f << 4) + lm;
            bf.h[0] = *(const v8h*)&Bs[buf][n * TK + kb];
            bf.h[1] = *(const v8h*)&Bs[buf][n * TK + kb + 16];
            acc[0][f] = __builtin_amdgcn_wmma_f32_16x16x32_f16(
                false, af[0].v, false, bf.v, (short)0, acc[0][f], false, false);
            acc[1][f] = __builtin_amdgcn_wmma_f32_16x16x32_f16(
                false, af[1].v, false, bf.v, (short)0, acc[1][f], false, false);
        }
    };

    // ---- software pipeline over K slabs, double-buffered LDS ----
    const int nSlab = K / TK;
    loadG();
    storeL(0);
    __syncthreads();
    int buf = 0;
    for (int s = 0; s < nSlab; ++s) {
        if (s + 1 < nSlab) loadG();       // next-slab HBM loads issue first
        compute(buf);                      // 12 ds_load_b128 + 8 v_wmma
        if (s + 1 < nSlab) storeL(buf ^ 1);
        __syncthreads();
        buf ^= 1;
    }

    // ---- epilogue (C layout: VGPR r -> M = r + 8*(lane>=16), N = lane&15) ----
    #pragma unroll
    for (int f = 0; f < 4; ++f) {
        int n = tileN + (waveN << 6) + (f << 4) + lm;
        float bvv = bias ? bias[n] : 0.0f;
        #pragma unroll
        for (int mi = 0; mi < 2; ++mi) {
            #pragma unroll
            for (int r = 0; r < 8; ++r) {
                int m = tileM + (waveM << 5) + (mi << 4) + r + ((lane >> 4) << 3);
                float v = acc[mi][f][r] + bvv;
                size_t off = (size_t)m * N + n;
                if (epilogue == 1)      C[off] = gelu_f(v);
                else if (epilogue == 2) { C[off] = v; C2[off] = gelu_f(v); }
                else                    C[off] = v;
            }
        }
    }
}

// ---------------------------------------------------------------------------
// Glue kernels
// ---------------------------------------------------------------------------
__global__ void prep_kernel(const float* __restrict__ W3, float* __restrict__ w3sum,
                            float* __restrict__ scal) {
    int i = blockIdx.x * blockDim.x + threadIdx.x;
    if (i < 512) {
        float s = 0.0f;
        for (int j = 0; j < 1024; ++j) s += W3[i * 1024 + j];
        w3sum[i] = s;
    }
    if (i < 16) scal[i] = 0.0f;
}

__global__ void concat_kernel(const float* __restrict__ q, const float* __restrict__ p,
                              float* __restrict__ qp, int n) {
    int idx = blockIdx.x * blockDim.x + threadIdx.x;
    if (idx >= n) return;
    int b = idx >> 11, c = idx & 2047;
    qp[idx] = (c < 1024) ? q[(size_t)b * 1024 + c] : p[(size_t)b * 1024 + (c - 1024)];
}

__global__ void dz2_kernel(const float* __restrict__ z2, const float* __restrict__ w3sum,
                           float* __restrict__ dz2, int n) {
    int idx = blockIdx.x * blockDim.x + threadIdx.x;
    if (idx >= n) return;
    dz2[idx] = w3sum[idx & 511] * dgelu_f(z2[idx]);
}

__global__ void dz1_kernel(float* __restrict__ dh1, const float* __restrict__ z1, int n) {
    int idx = blockIdx.x * blockDim.x + threadIdx.x;
    if (idx >= n) return;
    dh1[idx] *= dgelu_f(z1[idx]);
}

__global__ void leap1_kernel(const float* __restrict__ p, const float* __restrict__ q,
                             const float* __restrict__ g, const float* __restrict__ dtp,
                             float* __restrict__ ph, float* __restrict__ qf, int n) {
    int idx = blockIdx.x * blockDim.x + threadIdx.x;
    if (idx >= n) return;
    int b = idx >> 10, c = idx & 1023;
    float dt = 1.0f / (1.0f + __expf(-dtp[0]));
    ph[idx] = p[idx] - 0.5f * dt * g[(size_t)b * 2048 + c];
    qf[idx] = q[idx] + dt * g[(size_t)b * 2048 + 1024 + c];
}

__global__ void leap2_kernel(const float* __restrict__ ph, const float* __restrict__ g,
                             const float* __restrict__ dtp, float* __restrict__ pf, int n) {
    int idx = blockIdx.x * blockDim.x + threadIdx.x;
    if (idx >= n) return;
    int b = idx >> 10, c = idx & 1023;
    float dt = 1.0f / (1.0f + __expf(-dtp[0]));
    pf[idx] = ph[idx] - 0.5f * dt * g[(size_t)b * 2048 + c];
}

__global__ void volume_kernel(const float* __restrict__ qf, const float* __restrict__ pf,
                              float* __restrict__ out, int nrows) {
    int wid  = (blockIdx.x * blockDim.x + threadIdx.x) >> 5;
    int lane = threadIdx.x & 31;
    if (wid >= nrows) return;
    const float* qr = qf + (size_t)wid * 1024;
    const float* pr = pf + (size_t)wid * 1024;
    float sq = 0.0f, sp = 0.0f;
    for (int i = lane; i < 1024; i += 32) { float a = qr[i]; sq += a * a; float b = pr[i]; sp += b * b; }
    sq = wave_sum(sq); sp = wave_sum(sp);
    if (lane == 0) out[wid] = sqrtf(sq) * sqrtf(sp);
}

// LayerNorm over rows of 512; optional fused GELU. One wave per row.
__global__ void ln_kernel(const float* __restrict__ x, const float* __restrict__ g,
                          const float* __restrict__ b, float* __restrict__ y,
                          int nrows, int dogelu) {
    int wid  = (blockIdx.x * blockDim.x + threadIdx.x) >> 5;
    int lane = threadIdx.x & 31;
    if (wid >= nrows) return;
    const float* xr = x + (size_t)wid * 512;
    float v[16], s = 0.0f, ss = 0.0f;
    #pragma unroll
    for (int t = 0; t < 16; ++t) { float a = xr[lane + 32 * t]; v[t] = a; s += a; ss += a * a; }
    s = wave_sum(s); ss = wave_sum(ss);
    float mean = s * (1.0f / 512.0f);
    float var  = ss * (1.0f / 512.0f) - mean * mean;
    float inv  = 1.0f / sqrtf(var + 1e-5f);
    float* yr  = y + (size_t)wid * 512;
    #pragma unroll
    for (int t = 0; t < 16; ++t) {
        int i = lane + 32 * t;
        float o = (v[t] - mean) * inv * g[i] + b[i];
        yr[i] = dogelu ? gelu_f(o) : o;
    }
}

__global__ void isfp_kernel(const float* __restrict__ xs, const float* __restrict__ qf,
                            float* __restrict__ cnt, int nrows) {
    int wid  = (blockIdx.x * blockDim.x + threadIdx.x) >> 5;
    int lane = threadIdx.x & 31;
    if (wid >= nrows) return;
    float s = 0.0f;
    for (int i = lane; i < 512; i += 32) {
        float d = xs[(size_t)wid * 512 + i] - qf[(size_t)wid * 1024 + i];
        s += d * d;
    }
    s = wave_sum(s);
    if (lane == 0 && sqrtf(s) < 0.1f) atomicAdd(cnt, 1.0f);
}

__global__ void add_kernel(const float* __restrict__ a, const float* __restrict__ b,
                           float* __restrict__ c, int n) {
    int idx = blockIdx.x * blockDim.x + threadIdx.x;
    if (idx < n) c[idx] = a[idx] + b[idx];
}

__global__ void causal_kernel(const float* __restrict__ i0, const float* __restrict__ i1,
                              const float* __restrict__ i2, float* __restrict__ scal, int n) {
    int idx = blockIdx.x * blockDim.x + threadIdx.x;
    float a = 0.0f, b = 0.0f, c = 0.0f;
    if (idx < n) { a = i0[idx]; b = i1[idx]; c = i2[idx]; }
    float d01 = a - b, d02 = a - c, d12 = b - c;
    float s01 = wave_sum(d01 * d01);
    float s02 = wave_sum(d02 * d02);
    float s12 = wave_sum(d12 * d12);
    if ((threadIdx.x & 31) == 0) {
        atomicAdd(&scal[1], s01); atomicAdd(&scal[2], s02); atomicAdd(&scal[3], s12);
    }
}

__global__ void betti_kernel(const float* __restrict__ fe, const float* __restrict__ W,
                             const float* __restrict__ bv, float* __restrict__ out, int nrows) {
    int wid  = (blockIdx.x * blockDim.x + threadIdx.x) >> 5;
    int lane = threadIdx.x & 31;
    if (wid >= nrows) return;
    float l0 = 0.0f, l1 = 0.0f, l2 = 0.0f;
    for (int i = lane; i < 512; i += 32) {
        float f = fe[(size_t)wid * 512 + i];
        l0 += f * W[i * 3 + 0]; l1 += f * W[i * 3 + 1]; l2 += f * W[i * 3 + 2];
    }
    l0 = wave_sum(l0); l1 = wave_sum(l1); l2 = wave_sum(l2);
    if (lane == 0) {
        l0 += bv[0]; l1 += bv[1]; l2 += bv[2];
        float mx = fmaxf(l0, fmaxf(l1, l2));
        float e0 = __expf(l0 - mx), e1 = __expf(l1 - mx), e2 = __expf(l2 - mx);
        float s = e0 + e1 + e2;
        out[(size_t)wid * 3 + 0] = e0 / s;
        out[(size_t)wid * 3 + 1] = e1 / s;
        out[(size_t)wid * 3 + 2] = e2 / s;
    }
}

__global__ void lifetime_kernel(const float* __restrict__ ps, float* __restrict__ out, int nrows) {
    int wid  = (blockIdx.x * blockDim.x + threadIdx.x) >> 5;
    int lane = threadIdx.x & 31;
    if (wid >= nrows) return;
    float s = 0.0f;
    for (int i = lane; i < 256; i += 32) { float a = ps[(size_t)wid * 256 + i]; s += a * a; }
    s = wave_sum(s);
    if (lane == 0) out[wid] = sqrtf(s);
}

__global__ void finalize_kernel(const float* __restrict__ scal, float* __restrict__ isfp_out,
                                float* __restrict__ causal_out, int nrows) {
    if (threadIdx.x == 0 && blockIdx.x == 0) {
        isfp_out[0]   = scal[0] / (float)nrows;
        causal_out[0] = (sqrtf(scal[1]) + sqrtf(scal[2]) + sqrtf(scal[3])) / 3.0f;
    }
}

// ---------------------------------------------------------------------------
// Host orchestration
// ---------------------------------------------------------------------------
extern "C" void kernel_launch(void* const* d_in, const int* in_sizes, int n_in,
                              void* d_out, int out_size, void* d_ws, size_t ws_size,
                              hipStream_t stream) {
    const float* x    = (const float*)d_in[0];
    const float* p    = (const float*)d_in[1];
    const float* in_W = (const float*)d_in[2];
    const float* in_b = (const float*)d_in[3];
    const float* hW1  = (const float*)d_in[4];
    const float* hb1  = (const float*)d_in[5];
    const float* hW2  = (const float*)d_in[6];
    const float* hb2  = (const float*)d_in[7];
    const float* hW3  = (const float*)d_in[8];
    const float* dtp  = (const float*)d_in[10];
    const float* rg_cW = (const float*)d_in[11];
    const float* rg_cb = (const float*)d_in[12];
    const float* rg_lng = (const float*)d_in[13];
    const float* rg_lnb = (const float*)d_in[14];
    const float* rg_pW = (const float*)d_in[15];
    const float* rg_pb = (const float*)d_in[16];
    const float* rg_sW = (const float*)d_in[17];
    const float* rg_sb = (const float*)d_in[18];
    const float* ivW1 = (const float*)d_in[19];   // 3 x 512 x 256
    const float* ivb1 = (const float*)d_in[20];   // 3 x 256
    const float* ivW2 = (const float*)d_in[21];   // 3 x 256 x 512
    const float* ivb2 = (const float*)d_in[22];   // 3 x 512
    const float* cfW  = (const float*)d_in[23];
    const float* cfb  = (const float*)d_in[24];
    const float* tpW  = (const float*)d_in[25];
    const float* tpb  = (const float*)d_in[26];
    const float* tpg  = (const float*)d_in[27];
    const float* tpbe = (const float*)d_in[28];
    const float* btW  = (const float*)d_in[29];
    const float* btb  = (const float*)d_in[30];
    const float* pdW  = (const float*)d_in[31];
    const float* pdb  = (const float*)d_in[32];
    const float* clsW = (const float*)d_in[33];
    const float* clsb = (const float*)d_in[34];

    const size_t B = B_ROWS;
    float* w = (float*)d_ws;

    // Workspace layout (floats). Total ~88.08M floats (~353 MB).
    float* W3S = w;            // 512
    float* SC  = w + 512;      // 16 scalar accumulators
    size_t off = 1024;
    float* Q   = w + off; off += B * 1024;
    float* QP  = w + off; off += B * 2048;
    float* Z1  = w + off; off += B * 512;
    float* H1  = w + off; off += B * 512;
    float* Z2  = w + off; off += B * 512;
    float* DZ2 = w + off; off += B * 512;
    float* DH1 = w + off; off += B * 512;
    float* G   = w + off; off += B * 2048;
    float* PH  = w + off; off += B * 1024;
    float* QF  = w + off; off += B * 1024;
    float* PF  = w + off; off += B * 1024;
    // Aliases (regions free after the Hamiltonian/leapfrog stages):
    float* XT = Z1;  float* XC = H1;  float* XP = Z2;  float* XS = DZ2;  float* CONF = DH1;
    float* T0  = G;
    float* HH  = G + B * 512;
    float* IV0 = G + B * 512 + B * 256;
    float* IV1 = IV0 + B * 512;
    float* IV2 = QP;
    float* TT  = QP + B * 512;
    float* FE  = QP + 2 * B * 512;
    float* PS  = QP + 3 * B * 512;

    float* out      = (float*)d_out;
    float* OUT_MAIN = out;                       // B*1024
    float* OUT_VOL  = out + B * 1024;            // B
    float* OUT_ISFP = out + B * 1024 + B;        // 1
    float* OUT_CAU  = OUT_ISFP + 1;              // 1
    float* OUT_BET  = OUT_CAU + 1;               // B*3
    float* OUT_LIFE = OUT_BET + B * 3;           // B

    dim3 blk256(256);
    auto gemm = [&](const float* A, const float* Bm, const float* bias, float* C, float* C2,
                    int M, int N, int K, int transB, int ep) {
        dim3 grid(N / TN, M / TM);
        gemm_wmma_kernel<<<grid, blk256, 0, stream>>>(A, Bm, bias, C, C2, M, N, K, transB, ep);
    };
    const int EW = 256;
    auto nb = [&](size_t n) { return (unsigned)((n + EW - 1) / EW); };
    const unsigned rowsBlocks = (unsigned)(B / 8);   // wave-per-row kernels, 8 waves/block

    // --- prep: w3sum = rowsum(ham_W3); zero scalar accumulators ---
    prep_kernel<<<2, 256, 0, stream>>>(hW3, W3S, SC);

    // --- input linear: q = x @ in_W + in_b ---
    gemm(x, in_W, in_b, Q, nullptr, B, 1024, 2048, 0, 0);

    // --- gradient of Hamiltonian at [q, p] ---
    concat_kernel<<<nb(B * 2048), EW, 0, stream>>>(Q, p, QP, (int)(B * 2048));
    gemm(QP, hW1, hb1, Z1, H1, B, 512, 2048, 0, 2);          // z1 + h1=gelu(z1)
    gemm(H1, hW2, hb2, Z2, nullptr, B, 512, 512, 0, 0);      // z2
    dz2_kernel<<<nb(B * 512), EW, 0, stream>>>(Z2, W3S, DZ2, (int)(B * 512));
    gemm(DZ2, hW2, nullptr, DH1, nullptr, B, 512, 512, 1, 0);   // dz2 @ W2^T
    dz1_kernel<<<nb(B * 512), EW, 0, stream>>>(DH1, Z1, (int)(B * 512));
    gemm(DH1, hW1, nullptr, G, nullptr, B, 2048, 512, 1, 0);    // dz1 @ W1^T -> g
    leap1_kernel<<<nb(B * 1024), EW, 0, stream>>>(p, Q, G, dtp, PH, QF, (int)(B * 1024));

    // --- gradient at [q_full, p_half] ---
    concat_kernel<<<nb(B * 2048), EW, 0, stream>>>(QF, PH, QP, (int)(B * 2048));
    gemm(QP, hW1, hb1, Z1, H1, B, 512, 2048, 0, 2);
    gemm(H1, hW2, hb2, Z2, nullptr, B, 512, 512, 0, 0);
    dz2_kernel<<<nb(B * 512), EW, 0, stream>>>(Z2, W3S, DZ2, (int)(B * 512));
    gemm(DZ2, hW2, nullptr, DH1, nullptr, B, 512, 512, 1, 0);
    dz1_kernel<<<nb(B * 512), EW, 0, stream>>>(DH1, Z1, (int)(B * 512));
    gemm(DH1, hW1, nullptr, G, nullptr, B, 2048, 512, 1, 0);
    leap2_kernel<<<nb(B * 1024), EW, 0, stream>>>(PH, G, dtp, PF, (int)(B * 1024));

    volume_kernel<<<rowsBlocks, 256, 0, stream>>>(QF, PF, OUT_VOL, (int)B);

    // --- recursive refinement loop (5 iterations; pad region is all-zero so
    //     iterations 2-5 use K=512 with the first 512 rows of rg_cW) ---
    for (int it = 0; it < 5; ++it) {
        if (it == 0) gemm(QF, rg_cW, rg_cb, XT, nullptr, B, 512, 1024, 0, 0);
        else         gemm(XS, rg_cW, rg_cb, XT, nullptr, B, 512, 512, 0, 0);
        ln_kernel<<<rowsBlocks, 256, 0, stream>>>(XT, rg_lng, rg_lnb, XC, (int)B, 1);
        gemm(XC, rg_pW, rg_pb, XP, nullptr, B, 512, 512, 0, 0);
        gemm(XP, rg_sW, rg_sb, XS, nullptr, B, 512, 512, 0, 0);
    }

    isfp_kernel<<<rowsBlocks, 256, 0, stream>>>(XS, QF, &SC[0], (int)B);

    // --- intervention branches ---
    gemm(XS, cfW, cfb, CONF, nullptr, B, 512, 512, 0, 0);
    add_kernel<<<nb(B * 512), EW, 0, stream>>>(XS, CONF, T0, (int)(B * 512));
    gemm(T0, ivW1, ivb1, HH, nullptr, B, 256, 512, 0, 1);
    gemm(HH, ivW2, ivb2, IV0, nullptr, B, 512, 256, 0, 0);
    gemm(XS, ivW1 + 512 * 256, ivb1 + 256, HH, nullptr, B, 256, 512, 0, 1);
    gemm(HH, ivW2 + 256 * 512, ivb2 + 512, IV1, nullptr, B, 512, 256, 0, 0);
    gemm(XS, ivW1 + 2 * 512 * 256, ivb1 + 2 * 256, HH, nullptr, B, 256, 512, 0, 1);
    gemm(HH, ivW2 + 2 * 256 * 512, ivb2 + 2 * 512, IV2, nullptr, B, 512, 256, 0, 0);
    causal_kernel<<<nb(B * 512), EW, 0, stream>>>(IV0, IV1, IV2, SC, (int)(B * 512));

    // --- heads ---
    gemm(IV0, tpW, tpb, TT, nullptr, B, 512, 512, 0, 0);
    ln_kernel<<<rowsBlocks, 256, 0, stream>>>(TT, tpg, tpbe, FE, (int)B, 0);
    betti_kernel<<<rowsBlocks, 256, 0, stream>>>(FE, btW, btb, OUT_BET, (int)B);
    gemm(FE, pdW, pdb, PS, nullptr, B, 256, 512, 0, 0);
    lifetime_kernel<<<rowsBlocks, 256, 0, stream>>>(PS, OUT_LIFE, (int)B);
    gemm(FE, clsW, clsb, OUT_MAIN, nullptr, B, 1024, 512, 0, 0);

    finalize_kernel<<<1, 32, 0, stream>>>(SC, OUT_ISFP, OUT_CAU, (int)B);
}
